// CompGCNClassifier_22393959481892
// MI455X (gfx1250) — compile-verified, matrix operand order checked by reading, not run on hardware
//
#include <hip/hip_runtime.h>
#include <hip/hip_bf16.h>
#include <math.h>

#define NN 100000
#define NE 1000000
#define D 64
#define DIN 192
#define NCLS 16

typedef __bf16 bf16_t;
typedef bf16_t v16bf __attribute__((ext_vector_type(16)));
typedef bf16_t v8bf  __attribute__((ext_vector_type(8)));
typedef float  v8f   __attribute__((ext_vector_type(8)));

// ---------------------------------------------------------------------------
// Fragment layout for v_wmma_f32_16x16x32_bf16 (ISA 7.12.2):
// lane L holds row/col M|N = L&15; elements 0-7 -> K = kb..kb+7,
// elements 8-15 -> K = kb+16..kb+23, where kb = kt*32 + (L&16 ? 8 : 0).
// ---------------------------------------------------------------------------
__device__ __forceinline__ v16bf load_frag_f32(const float* __restrict__ row, int kbase) {
  const float4* q0 = reinterpret_cast<const float4*>(row + kbase);
  const float4* q1 = reinterpret_cast<const float4*>(row + kbase + 16);
  float4 x0 = q0[0], x1 = q0[1], x2 = q1[0], x3 = q1[1];
  v16bf a;
  a[0]  = (bf16_t)x0.x; a[1]  = (bf16_t)x0.y; a[2]  = (bf16_t)x0.z; a[3]  = (bf16_t)x0.w;
  a[4]  = (bf16_t)x1.x; a[5]  = (bf16_t)x1.y; a[6]  = (bf16_t)x1.z; a[7]  = (bf16_t)x1.w;
  a[8]  = (bf16_t)x2.x; a[9]  = (bf16_t)x2.y; a[10] = (bf16_t)x2.z; a[11] = (bf16_t)x2.w;
  a[12] = (bf16_t)x3.x; a[13] = (bf16_t)x3.y; a[14] = (bf16_t)x3.z; a[15] = (bf16_t)x3.w;
  return a;
}

// B fragments from pre-converted bf16 weights: two 16-byte loads, no cvt.
__device__ __forceinline__ v16bf load_frag_bf16(const bf16_t* __restrict__ row, int kbase) {
  v8bf lo = *reinterpret_cast<const v8bf*>(row + kbase);
  v8bf hi = *reinterpret_cast<const v8bf*>(row + kbase + 16);
  return __builtin_shufflevector(lo, hi, 0, 1, 2, 3, 4, 5, 6, 7,
                                 8, 9, 10, 11, 12, 13, 14, 15);
}

__device__ __forceinline__ v8f wmma_bf16(v16bf a, v16bf b, v8f c) {
  return __builtin_amdgcn_wmma_f32_16x16x32_bf16(false, a, false, b, (short)0, c, false, false);
}

// HW-native L2 float atomic (no return -> STOREcnt), device scope.
__device__ __forceinline__ void atomic_add_f32_dev(float* p, float v) {
  asm volatile("global_atomic_add_f32 %0, %1, off scope:SCOPE_DEV"
               :: "v"(p), "v"(v) : "memory");
}

// ---- fast transcendentals: hardware v_rcp_f32 / v_exp_f32 (/ v_tanh if present)
__device__ __forceinline__ float rcp_f(float x) {
#if __has_builtin(__builtin_amdgcn_rcpf)
  return __builtin_amdgcn_rcpf(x);
#else
  return 1.0f / x;
#endif
}
__device__ __forceinline__ float exp2_f(float x) {
#if __has_builtin(__builtin_amdgcn_exp2f)
  return __builtin_amdgcn_exp2f(x);
#else
  return exp2f(x);
#endif
}
__device__ __forceinline__ float tanh_f(float x) {
#if __has_builtin(__builtin_amdgcn_tanhf)
  return __builtin_amdgcn_tanhf(x);
#else
  // tanh(x) = 1 - 2/(1 + e^{2x}) ; e^{2x} = 2^{2*log2(e)*x}
  return 1.0f - 2.0f * rcp_f(1.0f + exp2_f(x * 2.8853900817779268f));
#endif
}
__device__ __forceinline__ float sigmoid_f(float x) {
  return rcp_f(1.0f + exp2_f(-x * 1.4426950408889634f));
}
__device__ __forceinline__ float gelu_f(float x) {
  float x3 = x * x * x;
  return 0.5f * x * (1.0f + tanh_f(0.7978845608028654f * (x + 0.044715f * x3)));
}

// ---------------------------------------------------------------------------
__global__ void zero_ws_kernel(float4* __restrict__ p, int n4) {
  int i = blockIdx.x * blockDim.x + threadIdx.x;
  if (i < n4) p[i] = make_float4(0.f, 0.f, 0.f, 0.f);
}

__global__ void cvt_weights_kernel(const float* __restrict__ src,
                                   bf16_t* __restrict__ dst, int n) {
  int i = blockIdx.x * blockDim.x + threadIdx.x;
  if (i < n) dst[i] = (bf16_t)src[i];
}

// msg = GELU([nf[src] | nf[tgt] | ef] @ Wm^T + bm) ; agg[tgt] += msg  (fused)
__global__ __launch_bounds__(256) void msg_scatter_kernel(
    const float* __restrict__ nf, const int* __restrict__ ei,
    const float* __restrict__ ef, const bf16_t* __restrict__ Wm,
    const float* __restrict__ bm, float* __restrict__ agg) {
  const int lane = threadIdx.x & 31;
  const int tile = blockIdx.x * 8 + (threadIdx.x >> 5);
  if (tile >= NE / 16) return;
  const int ln   = lane & 15;
  const int koff = (lane & 16) ? 8 : 0;
  const int hi8  = koff;
  const int base = tile * 16;
  const int e = base + ln;
  const float* srow = nf + (size_t)ei[e] * D;
  const float* trow = nf + (size_t)ei[NE + e] * D;
  const float* erow = ef + (size_t)e * D;

  v8f zero = {};
  v8f acc[4] = {zero, zero, zero, zero};
#pragma unroll
  for (int kt = 0; kt < 6; ++kt) {
    const float* row = (kt < 2) ? srow : (kt < 4) ? trow : erow;
    v16bf a = load_frag_f32(row, (kt & 1) * 32 + koff);
#pragma unroll
    for (int nt = 0; nt < 4; ++nt) {
      v16bf b = load_frag_bf16(Wm + (size_t)(nt * 16 + ln) * DIN, kt * 32 + koff);
      acc[nt] = wmma_bf16(a, b, acc[nt]);
    }
  }
  int trows[8];
#pragma unroll
  for (int r = 0; r < 8; ++r) trows[r] = ei[NE + base + r + hi8];
#pragma unroll
  for (int nt = 0; nt < 4; ++nt) {
    int o = nt * 16 + ln;
    float bias = bm[o];
#pragma unroll
    for (int r = 0; r < 8; ++r) {
      float g = gelu_f(acc[nt][r] + bias);
      atomic_add_f32_dev(agg + (size_t)trows[r] * D + o, g);
    }
  }
}

// GRU cell: hbuf holds agg on entry, h_new on exit (in-place, per-tile safe)
__global__ __launch_bounds__(256) void gru_kernel(
    const float* __restrict__ nf, const bf16_t* __restrict__ Wih,
    const float* __restrict__ bih, const bf16_t* __restrict__ Whh,
    const float* __restrict__ bhh, float* __restrict__ hbuf) {
  const int lane = threadIdx.x & 31;
  const int tile = blockIdx.x * 8 + (threadIdx.x >> 5);
  if (tile >= NN / 16) return;
  const int ln = lane & 15;
  const int koff = (lane & 16) ? 8 : 0;
  const int hi8 = koff;
  const int base = tile * 16;
  const float* arow = hbuf + (size_t)(base + ln) * D;
  const float* hrow = nf + (size_t)(base + ln) * D;

  v16bf a_ag[2], a_h[2];
#pragma unroll
  for (int kt = 0; kt < 2; ++kt) {
    a_ag[kt] = load_frag_f32(arow, kt * 32 + koff);
    a_h[kt]  = load_frag_f32(hrow, kt * 32 + koff);
  }
  v8f zero = {};
#pragma unroll
  for (int dt = 0; dt < 4; ++dt) {
    v8f ir = zero, iz = zero, inn = zero, hr = zero, hz = zero, hn = zero;
#pragma unroll
    for (int kt = 0; kt < 2; ++kt) {
      int kb = kt * 32 + koff;
      const size_t jr = (size_t)(dt * 16 + ln) * D;
      ir  = wmma_bf16(a_ag[kt], load_frag_bf16(Wih + jr, kb), ir);
      iz  = wmma_bf16(a_ag[kt], load_frag_bf16(Wih + jr + (size_t)64 * D, kb), iz);
      inn = wmma_bf16(a_ag[kt], load_frag_bf16(Wih + jr + (size_t)128 * D, kb), inn);
      hr  = wmma_bf16(a_h[kt],  load_frag_bf16(Whh + jr, kb), hr);
      hz  = wmma_bf16(a_h[kt],  load_frag_bf16(Whh + jr + (size_t)64 * D, kb), hz);
      hn  = wmma_bf16(a_h[kt],  load_frag_bf16(Whh + jr + (size_t)128 * D, kb), hn);
    }
    int j = dt * 16 + ln;
    float bir = bih[j], biz = bih[64 + j], bin = bih[128 + j];
    float bhr = bhh[j], bhz = bhh[64 + j], bhn = bhh[128 + j];
#pragma unroll
    for (int r = 0; r < 8; ++r) {
      int m = base + r + hi8;
      float rg = sigmoid_f(ir[r] + bir + hr[r] + bhr);
      float zg = sigmoid_f(iz[r] + biz + hz[r] + bhz);
      float ng = tanh_f(inn[r] + bin + rg * (hn[r] + bhn));
      float hold = nf[(size_t)m * D + j];
      hbuf[(size_t)m * D + j] = (1.0f - zg) * ng + zg * hold;
    }
  }
}

// out = GELU([h[src] | h[tgt] | ef] @ Wc1^T + bc1) @ Wc2^T + bc2
__global__ __launch_bounds__(256) void cls_kernel(
    const float* __restrict__ hbuf, const int* __restrict__ ei,
    const float* __restrict__ ef, const bf16_t* __restrict__ Wc1,
    const float* __restrict__ bc1, const bf16_t* __restrict__ Wc2,
    const float* __restrict__ bc2, float* __restrict__ out) {
  __shared__ bf16_t stage[8][16 * 64];  // per-wave 16x64 bf16 re-layout buffer
  const int lane = threadIdx.x & 31;
  const int wv = threadIdx.x >> 5;
  const int tile = blockIdx.x * 8 + wv;
  if (tile >= NE / 16) return;
  const int ln = lane & 15;
  const int koff = (lane & 16) ? 8 : 0;
  const int hi8 = koff;
  const int base = tile * 16;
  const int e = base + ln;
  const float* srow = hbuf + (size_t)ei[e] * D;
  const float* trow = hbuf + (size_t)ei[NE + e] * D;
  const float* erow = ef + (size_t)e * D;

  v8f zero = {};
  v8f acc[4] = {zero, zero, zero, zero};
#pragma unroll
  for (int kt = 0; kt < 6; ++kt) {
    const float* row = (kt < 2) ? srow : (kt < 4) ? trow : erow;
    v16bf a = load_frag_f32(row, (kt & 1) * 32 + koff);
#pragma unroll
    for (int nt = 0; nt < 4; ++nt) {
      v16bf b = load_frag_bf16(Wc1 + (size_t)(nt * 16 + ln) * DIN, kt * 32 + koff);
      acc[nt] = wmma_bf16(a, b, acc[nt]);
    }
  }
  // bias + GELU, re-layout D (row=r+hi8, col=nt*16+ln) -> A via LDS (bf16)
#pragma unroll
  for (int nt = 0; nt < 4; ++nt) {
    float bias = bc1[nt * 16 + ln];
#pragma unroll
    for (int r = 0; r < 8; ++r)
      stage[wv][(r + hi8) * 64 + nt * 16 + ln] = (bf16_t)gelu_f(acc[nt][r] + bias);
  }
  asm volatile("s_wait_dscnt 0x0" ::: "memory");  // same-wave LDS store->load ordering

  v8f o2 = zero;
#pragma unroll
  for (int kt = 0; kt < 2; ++kt) {
    v16bf a2;
#pragma unroll
    for (int i = 0; i < 16; ++i) {
      int k = kt * 32 + ((i < 8) ? i : i + 8) + koff;
      a2[i] = stage[wv][ln * 64 + k];
    }
    v16bf b2 = load_frag_bf16(Wc2 + (size_t)ln * D, kt * 32 + koff);
    o2 = wmma_bf16(a2, b2, o2);
  }
  float bias2 = bc2[ln];
#pragma unroll
  for (int r = 0; r < 8; ++r)
    out[(size_t)(base + r + hi8) * NCLS + ln] = o2[r] + bias2;
}

// ---------------------------------------------------------------------------
extern "C" void kernel_launch(void* const* d_in, const int* in_sizes, int n_in,
                              void* d_out, int out_size, void* d_ws, size_t ws_size,
                              hipStream_t stream) {
  (void)in_sizes; (void)n_in; (void)out_size; (void)ws_size;
  const float* nf  = (const float*)d_in[0];
  const int*   ei  = (const int*)d_in[1];
  const float* ef  = (const float*)d_in[2];
  const float* Wm  = (const float*)d_in[3];
  const float* bm  = (const float*)d_in[4];
  const float* Wih = (const float*)d_in[5];
  const float* bih = (const float*)d_in[6];
  const float* Whh = (const float*)d_in[7];
  const float* bhh = (const float*)d_in[8];
  const float* Wc1 = (const float*)d_in[9];
  const float* bc1 = (const float*)d_in[10];
  const float* Wc2 = (const float*)d_in[11];
  const float* bc2 = (const float*)d_in[12];
  float* out = (float*)d_out;

  // ws layout: [agg/h : NN*D f32 = 25.6 MB][bf16 weights ~100 KB]
  float* agg = (float*)d_ws;
  bf16_t* wbase  = (bf16_t*)((char*)d_ws + (size_t)NN * D * sizeof(float));
  bf16_t* Wm_b   = wbase;                      // 64*192
  bf16_t* Wih_b  = Wm_b  + (size_t)D * DIN;    // 192*64
  bf16_t* Whh_b  = Wih_b + (size_t)3 * D * D;  // 192*64
  bf16_t* Wc1_b  = Whh_b + (size_t)3 * D * D;  // 64*192
  bf16_t* Wc2_b  = Wc1_b + (size_t)D * DIN;    // 16*64

  const int n4 = NN * D / 4;
  zero_ws_kernel<<<(n4 + 255) / 256, 256, 0, stream>>>((float4*)d_ws, n4);
  cvt_weights_kernel<<<(D * DIN + 255) / 256, 256, 0, stream>>>(Wm, Wm_b, D * DIN);
  cvt_weights_kernel<<<(3 * D * D + 255) / 256, 256, 0, stream>>>(Wih, Wih_b, 3 * D * D);
  cvt_weights_kernel<<<(3 * D * D + 255) / 256, 256, 0, stream>>>(Whh, Whh_b, 3 * D * D);
  cvt_weights_kernel<<<(D * DIN + 255) / 256, 256, 0, stream>>>(Wc1, Wc1_b, D * DIN);
  cvt_weights_kernel<<<(NCLS * D + 255) / 256, 256, 0, stream>>>(Wc2, Wc2_b, NCLS * D);

  const int etiles = NE / 16;  // 62500, exact
  const int ntiles = NN / 16;  // 6250, exact
  msg_scatter_kernel<<<(etiles + 7) / 8, 256, 0, stream>>>(nf, ei, ef, Wm_b, bm, agg);
  gru_kernel<<<(ntiles + 7) / 8, 256, 0, stream>>>(nf, Wih_b, bih, Whh_b, bhh, agg);
  cls_kernel<<<(etiles + 7) / 8, 256, 0, stream>>>(agg, ei, ef, Wc1_b, bc1, Wc2_b, bc2, out);
}
